// HMM_86260123174462
// MI455X (gfx1250) — matrix operand order repaired and checked by reading, not compiled, not run on hardware
//
#include <hip/hip_runtime.h>

// Problem constants (match the JAX reference)
#define H 512
#define V 50257
#define B 64
#define T 512

typedef __attribute__((ext_vector_type(16))) __bf16 v16bf;
typedef __attribute__((ext_vector_type(8)))  float  v8f;

struct __attribute__((aligned(16))) U128 { unsigned x, y, z, w; };
union BfVec { v16bf v; U128 q[2]; };

// float -> bf16 bits, round-to-nearest-even (inputs are positive finite probs)
__device__ __forceinline__ unsigned short f2bf(float f) {
    unsigned u = __builtin_bit_cast(unsigned, f);
    u += 0x7FFFu + ((u >> 16) & 1u);
    return (unsigned short)(u >> 16);
}

// ---------------------------------------------------------------------------
// One-time prep: W[i][j] = (alpha_exp[j][i]+1e-12) / (max_j alpha_exp[j][i]+1e-12)
// stored row-major (i-major) as bf16;  wmax[i] = log(max_j alpha_exp[j][i]+1e-12)
// One wave32 per output row i.
// ---------------------------------------------------------------------------
__global__ void prep_w_kernel(const float* __restrict__ alpha_exp,
                              unsigned short* __restrict__ Wb,
                              float* __restrict__ wmax) {
    int i = blockIdx.x * (blockDim.x >> 5) + (threadIdx.x >> 5);
    int lane = threadIdx.x & 31;
    if (i >= H) return;
    float m = 0.0f;
    for (int j = lane; j < H; j += 32)
        m = fmaxf(m, alpha_exp[(size_t)j * H + i]);
    for (int off = 16; off; off >>= 1)
        m = fmaxf(m, __shfl_xor(m, off, 32));
    float denom = m + 1e-12f;
    if (lane == 0) wmax[i] = logf(denom);
    float inv = 1.0f / denom;
    for (int j = lane; j < H; j += 32)
        Wb[(size_t)i * H + j] = f2bf((alpha_exp[(size_t)j * H + i] + 1e-12f) * inv);
}

// ---------------------------------------------------------------------------
// Persistent scan kernel: one workgroup per 16-batch tile runs ALL T-1 steps.
// alpha stays resident in LDS for the whole scan; only beta/W/ids are read
// from global (W is L2-resident: 512 KB bf16, re-streamed each step).
//
// Block: 512 threads = 16 waves. Wave w owns batch row w (phase 1) and
// i-tiles [w*32, w*32+16) / [w*32+16, w*32+32) (phase 2/3, dual WMMA chains).
// The beta gather for the step's tokens is issued BEFORE the K-loop so its
// scattered-load latency hides under the WMMA pipeline.
// ---------------------------------------------------------------------------
#define AS_STRIDE 516   // float stride: 4-bank skew per row -> conflict-free col writes
#define XS_STRIDE 520   // ushort stride: 16B aligned, 4-bank skew per row

__global__ void __launch_bounds__(512)
hmm_scan_kernel(const float* __restrict__ gamma_exp,
                const float* __restrict__ beta,
                const int*   __restrict__ ids,
                const unsigned short* __restrict__ Wb,
                const float* __restrict__ wmax,
                float* __restrict__ out) {
    __shared__ float          alphaS[16][AS_STRIDE];
    __shared__ unsigned short Xs[16][XS_STRIDE];
    __shared__ float          xmax_s[16];
    __shared__ int            tok_s[16];

    const int tid  = threadIdx.x;
    const int wv   = tid >> 5;          // wave id: batch row (phase1) / i-group (phase2)
    const int lane = tid & 31;
    const int half = lane >> 4;         // 0: lanes 0-15, 1: lanes 16-31
    const int l16  = lane & 15;
    const int mb   = blockIdx.x;        // batch tile
    const int b    = mb * 16 + wv;

    // Loop-invariant per-wave state for the two owned i-tiles
    const int iA = wv * 32 + l16;
    const int iB = wv * 32 + 16 + l16;
    const unsigned short* wrow0 = Wb + (size_t)iA * H;
    const unsigned short* wrow1 = Wb + (size_t)iB * H;
    const float  wmA   = wmax[iA];
    const float  wmB   = wmax[iB];
    const float* betaA = beta + (size_t)iA * V;
    const float* betaB = beta + (size_t)iB * V;
    const int*   idrow = ids + (size_t)b * T;

    // ---- init: alpha0[b][h] = log(gamma[h]) + beta[h][ids[b][0]]
    {
        int tok0 = idrow[0];
        for (int j = lane; j < H; j += 32)
            alphaS[wv][j] = logf(gamma_exp[j]) + beta[(size_t)j * V + tok0];
    }
    __syncthreads();

    for (int t = 1; t < T; ++t) {
        // ---- Phase 1: wave wv handles batch row wv. Row max via shuffles,
        //      then X = bf16(exp(alpha - max)) into LDS staging.
        float av[16];
        float m = -INFINITY;
        #pragma unroll
        for (int k = 0; k < 16; ++k) {
            av[k] = alphaS[wv][lane + 32 * k];
            m = fmaxf(m, av[k]);
        }
        #pragma unroll
        for (int off = 16; off; off >>= 1)
            m = fmaxf(m, __shfl_xor(m, off, 32));
        #pragma unroll
        for (int k = 0; k < 16; ++k)
            Xs[wv][lane + 32 * k] = f2bf(__expf(av[k] - m));
        if (lane == 0) {
            xmax_s[wv] = m;
            tok_s[wv]  = idrow[t];
        }
        __syncthreads();

        // ---- Prefetch the epilogue operands NOW: they depend only on the
        //      tokens (known after phase 1), not on the WMMA results, so the
        //      scattered beta gathers overlap the whole K-loop below.
        float xm8[8], bvA[8], bvB[8];
        #pragma unroll
        for (int v = 0; v < 8; ++v) {
            int r  = v + half * 8;
            int tk = tok_s[r];
            xm8[v] = xmax_s[r];
            bvA[v] = betaA[tk];
            bvB[v] = betaB[tk];
        }

        // ---- Phase 2: Z tiles via bf16 WMMA, two accumulation chains/wave.
        v8f c0 = {}, c1 = {};
        #pragma unroll 4
        for (int k0 = 0; k0 < H; k0 += 32) {
            // A tile (16x32): lane = M; lanes 0-15: K {k0..k0+15},
            //                           lanes 16-31: K {k0+16..k0+31}
            BfVec a, b0, b1;
            const unsigned short* pa = &Xs[l16][k0 + half * 16];
            a.q[0] = *(const U128*)(pa);
            a.q[1] = *(const U128*)(pa + 8);
            // B tiles (32x16): lane = N; contiguous 16-K run per half
            const unsigned short* p0 = wrow0 + k0 + half * 16;
            b0.q[0] = *(const U128*)(p0);
            b0.q[1] = *(const U128*)(p0 + 8);
            const unsigned short* p1 = wrow1 + k0 + half * 16;
            b1.q[0] = *(const U128*)(p1);
            b1.q[1] = *(const U128*)(p1 + 8);
            c0 = __builtin_amdgcn_wmma_f32_16x16x32_bf16(
                     false, a.v, false, b0.v, (short)0, c0, false, false);
            c1 = __builtin_amdgcn_wmma_f32_16x16x32_bf16(
                     false, a.v, false, b1.v, (short)0, c1, false, false);
        }

        // ---- Phase 3: alpha_new back into LDS.
        //      C/D layout: lane -> N (= i offset), vgpr v -> M = v + 8*half.
        #pragma unroll
        for (int v = 0; v < 8; ++v) {
            int r = v + half * 8;
            alphaS[r][iA] = __logf(c0[v]) + xm8[v] + wmA + bvA[v];
            alphaS[r][iB] = __logf(c1[v]) + xm8[v] + wmB + bvB[v];
        }
        __syncthreads();
    }

    // ---- write final alpha to output
    for (int j = lane; j < H; j += 32)
        out[(size_t)b * H + j] = alphaS[wv][j];
}

// ---------------------------------------------------------------------------
extern "C" void kernel_launch(void* const* d_in, const int* in_sizes, int n_in,
                              void* d_out, int out_size, void* d_ws, size_t ws_size,
                              hipStream_t stream) {
    const float* alpha_exp = (const float*)d_in[0];   // (H,H)
    const float* beta      = (const float*)d_in[1];   // (H,V)
    const float* gamma_exp = (const float*)d_in[2];   // (1,H)
    const int*   ids       = (const int*)d_in[3];     // (B,T)

    char* ws = (char*)d_ws;
    unsigned short* Wb = (unsigned short*)ws;               // H*H bf16 = 512 KB
    float* wmax = (float*)(ws + (size_t)H * H * 2);         // 2 KB

    prep_w_kernel<<<H / 8, 256, 0, stream>>>(alpha_exp, Wb, wmax);
    hmm_scan_kernel<<<B / 16, 512, 0, stream>>>(gamma_exp, beta, ids, Wb, wmax,
                                                (float*)d_out);
}
